// AttentionLayer_65833258713134
// MI455X (gfx1250) — matrix operand order repaired and checked by reading, not compiled, not run on hardware
//
#include <hip/hip_runtime.h>
#include <hip/hip_bf16.h>
#include <math.h>

#define H_DIM 1024
#define SLEN  2048
#define BATCH 64

typedef __attribute__((ext_vector_type(16))) __bf16 bf16x16;
typedef __attribute__((ext_vector_type(8)))  float  floatx8;

union Frag { unsigned u[8]; bf16x16 v; };

__device__ __forceinline__ unsigned pack_bf16x2(float lo, float hi) {
    union { float f; unsigned u; } a, b;
    a.f = lo; b.f = hi;
    unsigned ra = a.u + 0x7FFFu + ((a.u >> 16) & 1u);   // round-to-nearest-even
    unsigned rb = b.u + 0x7FFFu + ((b.u >> 16) & 1u);
    return (ra >> 16) | ((rb >> 16) << 16);
}

// branchless tanh: one v_exp_f32, no EXEC divergence; clamp avoids inf/inf
__device__ __forceinline__ float fast_tanh(float x) {
    x = fminf(fmaxf(x, -12.0f), 12.0f);
    float e = __expf(2.0f * x);
    return (e - 1.0f) * __frcp_rn(e + 1.0f);
}

// ---------------------------------------------------------------------------
// One-time: W0_top (k,n) f32 -> W0T (n,k) bf16 packed pairs (2 MB, L2-resident)
// ---------------------------------------------------------------------------
__global__ __launch_bounds__(256) void w0_transpose_bf16(const float* __restrict__ W0,
                                                         unsigned* __restrict__ w0t) {
    int idx = blockIdx.x * 256 + threadIdx.x;   // 1024*512 packed pairs
    int kk  = idx >> 10;                        // k pair index 0..511
    int n   = idx & 1023;
    float f0 = W0[(size_t)(2 * kk)     * H_DIM + n];
    float f1 = W0[(size_t)(2 * kk + 1) * H_DIM + n];
    w0t[(size_t)n * 512 + kk] = pack_bf16x2(f0, f1);
}

// ---------------------------------------------------------------------------
// g[b][n] = b0[n] + sum_k hidden[b][k] * W0[H+k][n]      (tiny GEMV per batch)
// ---------------------------------------------------------------------------
__global__ __launch_bounds__(256) void g_kernel(const float* __restrict__ hidden,
                                                const float* __restrict__ W0,
                                                const float* __restrict__ b0,
                                                float* __restrict__ g) {
    const int b = blockIdx.x;
    const int n = blockIdx.y * 256 + threadIdx.x;
    float acc = b0[n];
    const float* wp = W0 + (size_t)H_DIM * H_DIM + n;   // bottom half of W0
    const float* hp = hidden + b * H_DIM;
    for (int k = 0; k < H_DIM; ++k)
        acc += hp[k] * wp[(size_t)k * H_DIM];
    g[b * H_DIM + n] = acc;
}

// ---------------------------------------------------------------------------
// scores[b,s] = w1 . tanh(enc[b,s,:] @ W0_top + g[b,:])
// 1024 WGs; A tile (128 x 1024) converted to bf16 ONCE into 264 KB LDS;
// 64x64 accumulator tile per wave: 16 WMMAs per 8 ds_b128 + 8 global_b128.
// ---------------------------------------------------------------------------
#define A_STRIDE 516   // u32 per LDS row (1024 bf16 + pad); %64 avoids conflicts
__global__ __launch_bounds__(256) void attn_gemm_scores(
        const float* __restrict__ enc, const unsigned* __restrict__ w0t,
        const float* __restrict__ g,   const float* __restrict__ w1,
        float* __restrict__ scores) {
    __shared__ unsigned Asmem[128 * A_STRIDE];     // 128 rows x 1024 bf16 (k)
    __shared__ float    s_scores[128];

    const int tid     = threadIdx.x;
    const int rowBase = blockIdx.x * 128;          // flat (b,s) row
    const int lane    = tid & 31;
    const int w       = tid >> 5;                  // wave 0..7
    const int wm      = w & 1;                     // 2 waves along M (64 rows)
    const int wn      = w >> 1;                    // 4 waves along N (64 cols each)
    const int half    = lane >> 4;                 // lane group 0/1
    const int ln      = lane & 15;
    const int b       = rowBase / SLEN;            // constant per WG

    if (tid < 128) s_scores[tid] = 0.0f;

    // ---- stage + convert full A tile once: 65536 u32, 256 per thread ----
    {
        const float* src = enc + (size_t)rowBase * H_DIM;
        #pragma unroll 4
        for (int it = 0; it < 256; ++it) {
            int idx = it * 256 + tid;
            int row = idx >> 9;                    // 512 u32 per logical row
            int kk  = idx & 511;
            float f0 = src[(size_t)row * H_DIM + 2 * kk];
            float f1 = src[(size_t)row * H_DIM + 2 * kk + 1];
            Asmem[row * A_STRIDE + kk] = pack_bf16x2(f0, f1);
        }
    }
    __syncthreads();

    // fragment base addresses (u32 indices)
    const unsigned* abase[4];
    #pragma unroll
    for (int i = 0; i < 4; ++i)
        abase[i] = Asmem + (wm * 64 + i * 16 + ln) * A_STRIDE + half * 4;

    float rowsum[4][8];
    #pragma unroll
    for (int i = 0; i < 4; ++i)
        #pragma unroll
        for (int r = 0; r < 8; ++r) rowsum[i][r] = 0.0f;

    for (int nb = 0; nb < 4; ++nb) {               // N blocks of 256
        const int nBase = nb * 256;
        const unsigned* bbase[4];
        #pragma unroll
        for (int j = 0; j < 4; ++j)
            bbase[j] = w0t + (size_t)(nBase + wn * 64 + j * 16 + ln) * 512 + half * 4;

        floatx8 acc[4][4];
        #pragma unroll
        for (int i = 0; i < 4; ++i)
            #pragma unroll
            for (int j = 0; j < 4; ++j) acc[i][j] = {};

        for (int k0 = 0; k0 < H_DIM; k0 += 32) {   // 8 ds + 8 vmem + 16 wmma
            const int kk = k0 >> 1;
            Frag af[4], bf[4];
            #pragma unroll
            for (int i = 0; i < 4; ++i) {
                const uint4* p = (const uint4*)(abase[i] + kk);
                uint4 x0 = p[0], x1 = p[2];
                af[i].u[0]=x0.x; af[i].u[1]=x0.y; af[i].u[2]=x0.z; af[i].u[3]=x0.w;
                af[i].u[4]=x1.x; af[i].u[5]=x1.y; af[i].u[6]=x1.z; af[i].u[7]=x1.w;
            }
            #pragma unroll
            for (int j = 0; j < 4; ++j) {
                const uint4* p = (const uint4*)(bbase[j] + kk);
                uint4 x0 = p[0], x1 = p[2];
                bf[j].u[0]=x0.x; bf[j].u[1]=x0.y; bf[j].u[2]=x0.z; bf[j].u[3]=x0.w;
                bf[j].u[4]=x1.x; bf[j].u[5]=x1.y; bf[j].u[6]=x1.z; bf[j].u[7]=x1.w;
            }
            #pragma unroll
            for (int i = 0; i < 4; ++i)
                #pragma unroll
                for (int j = 0; j < 4; ++j)
                    acc[i][j] = __builtin_amdgcn_wmma_f32_16x16x32_bf16(
                        false, af[i].v, false, bf[j].v,
                        (short)0, acc[i][j], false, false);
        }

        // epilogue for this N block: tanh + dot with w1, kept in registers
        #pragma unroll
        for (int j = 0; j < 4; ++j) {
            int ng = nBase + wn * 64 + j * 16 + ln;
            float gv = g[b * H_DIM + ng];
            float wv = w1[ng];
            #pragma unroll
            for (int i = 0; i < 4; ++i)
                #pragma unroll
                for (int r = 0; r < 8; ++r)
                    rowsum[i][r] += fast_tanh(acc[i][j][r] + gv) * wv;
        }
    }

    // cross-lane (16-wide butterfly) then cross-wave (LDS) reduction
    #pragma unroll
    for (int i = 0; i < 4; ++i)
        #pragma unroll
        for (int r = 0; r < 8; ++r) {
            float v = rowsum[i][r];
            #pragma unroll
            for (int m = 1; m < 16; m <<= 1) v += __shfl_xor(v, m, 32);
            if (ln == 0)
                atomicAdd(&s_scores[wm * 64 + i * 16 + half * 8 + r], v);
        }
    __syncthreads();
    if (tid < 128) scores[rowBase + tid] = s_scores[tid];
}

// ---------------------------------------------------------------------------
// masked softmax over S per batch (in-place scores -> attn)
// ---------------------------------------------------------------------------
__global__ __launch_bounds__(256) void softmax_kernel(float* __restrict__ scores,
                                                      const unsigned char* __restrict__ mask,
                                                      const float* __restrict__ b1) {
    __shared__ float red[256];
    const int b = blockIdx.x, tid = threadIdx.x;
    const float b1v = b1[0];
    float vals[8];
    float lmax = -INFINITY;
    #pragma unroll
    for (int t = 0; t < 8; ++t) {
        int s = t * 256 + tid;
        float v = scores[b * SLEN + s] + b1v;
        if (!mask[b * SLEN + s]) v = -INFINITY;
        vals[t] = v;
        lmax = fmaxf(lmax, v);
    }
    red[tid] = lmax; __syncthreads();
    for (int o = 128; o > 0; o >>= 1) {
        if (tid < o) red[tid] = fmaxf(red[tid], red[tid + o]);
        __syncthreads();
    }
    const float M = red[0]; __syncthreads();
    float lsum = 0.0f;
    #pragma unroll
    for (int t = 0; t < 8; ++t) { float e = __expf(vals[t] - M); vals[t] = e; lsum += e; }
    red[tid] = lsum; __syncthreads();
    for (int o = 128; o > 0; o >>= 1) {
        if (tid < o) red[tid] += red[tid + o];
        __syncthreads();
    }
    const float inv = 1.0f / red[0];
    #pragma unroll
    for (int t = 0; t < 8; ++t) scores[b * SLEN + t * 256 + tid] = vals[t] * inv;
}

// ---------------------------------------------------------------------------
// new_hidden[b][h] = sum_s attn[b][s] * enc[b][s][h]   (bandwidth bound)
// ---------------------------------------------------------------------------
__global__ __launch_bounds__(256) void attn_context(const float* __restrict__ enc,
                                                    const float* __restrict__ attn,
                                                    float* __restrict__ out) {
    const int b = blockIdx.x;
    const int h = blockIdx.y * 256 + threadIdx.x;
    const float* e = enc + (size_t)b * SLEN * H_DIM + h;
    const float* a = attn + b * SLEN;
    float acc = 0.0f;
    #pragma unroll 4
    for (int s = 0; s < SLEN; ++s)
        acc += a[s] * e[(size_t)s * H_DIM];
    out[b * H_DIM + h] = acc;
}

// ---------------------------------------------------------------------------
extern "C" void kernel_launch(void* const* d_in, const int* in_sizes, int n_in,
                              void* d_out, int out_size, void* d_ws, size_t ws_size,
                              hipStream_t stream) {
    const float*         hidden = (const float*)d_in[0];
    const float*         enc    = (const float*)d_in[1];
    const unsigned char* mask   = (const unsigned char*)d_in[2];
    const float*         W0     = (const float*)d_in[3];
    const float*         b0     = (const float*)d_in[4];
    const float*         w1     = (const float*)d_in[5];
    const float*         b1     = (const float*)d_in[6];
    float* out = (float*)d_out;

    float*    scores = (float*)d_ws;                    // B*S floats (-> attn)
    float*    g      = scores + BATCH * SLEN;           // B*H floats
    unsigned* w0t    = (unsigned*)(g + BATCH * H_DIM);  // 1024*512 u32 (bf16 W0T)

    w0_transpose_bf16<<<dim3((H_DIM * 512) / 256), 256, 0, stream>>>(W0, w0t);
    g_kernel<<<dim3(BATCH, H_DIM / 256), 256, 0, stream>>>(hidden, W0, b0, g);
    attn_gemm_scores<<<dim3((BATCH * SLEN) / 128), 256, 0, stream>>>(enc, w0t, g, w1, scores);
    softmax_kernel<<<dim3(BATCH), 256, 0, stream>>>(scores, mask, b1);
    attn_context<<<dim3(BATCH, H_DIM / 256), 256, 0, stream>>>(enc, scores, out);
}
